// PerceiverSelfAttention_33904471834850
// MI455X (gfx1250) — matrix operand order repaired and checked by reading, not compiled, not run on hardware
//
#include <hip/hip_runtime.h>
#include <hip/hip_bf16.h>

typedef __attribute__((ext_vector_type(16))) _Float16 v16h;
typedef __attribute__((ext_vector_type(8)))  _Float16 v8h;
typedef __attribute__((ext_vector_type(8)))  float    v8f;

#define BB 4
#define SS 2048
#define DD 1024
#define HH 16
#define DH 64
#define NTOK (BB * SS)   // 8192

static __device__ __forceinline__ v16h cat8(v8h lo, v8h hi) {
    return __builtin_shufflevector(lo, hi, 0,1,2,3,4,5,6,7,8,9,10,11,12,13,14,15);
}

// 16-byte global -> LDS async copy (tracked by ASYNCcnt).
// vdst = wave-relative LDS byte offset (low 32 bits of flat shared address),
// vaddr = 64-bit global address. Inline asm: the clang builtin exists but its
// pointer params use the __device__/__shared__ LangAS, unspellable from HIP.
static __device__ __forceinline__ void cp_b128_to_lds(const _Float16* g, _Float16* l) {
    const unsigned int       ldsoff = (unsigned int)(uintptr_t)l;
    const unsigned long long gaddr  = (unsigned long long)(uintptr_t)g;
    asm volatile("global_load_async_to_lds_b128 %0, %1, off"
                 :: "v"(ldsoff), "v"(gaddr)
                 : "memory");
}
static __device__ __forceinline__ void wait_async_le(int pending) {
    if (pending) asm volatile("s_wait_asynccnt 0x4" ::: "memory");
    else         asm volatile("s_wait_asynccnt 0x0" ::: "memory");
}

// ---------------------------------------------------------------------------
// Kernel 1: LayerNorm (f32 in) -> f16 normalized activations
// ---------------------------------------------------------------------------
__global__ __launch_bounds__(256) void ln_kernel(const float* __restrict__ x,
                                                 const float* __restrict__ gamma,
                                                 const float* __restrict__ beta,
                                                 _Float16* __restrict__ xn) {
    __shared__ float sred[256];
    const int t   = blockIdx.x;
    const int tid = threadIdx.x;
    const float* row = x + (size_t)t * DD;

    float v[4];
    float s = 0.f;
#pragma unroll
    for (int i = 0; i < 4; ++i) { v[i] = row[tid + 256 * i]; s += v[i]; }
    sred[tid] = s;
    __syncthreads();
    for (int o = 128; o > 0; o >>= 1) {
        if (tid < o) sred[tid] += sred[tid + o];
        __syncthreads();
    }
    const float mu = sred[0] * (1.0f / DD);
    __syncthreads();

    float s2 = 0.f;
#pragma unroll
    for (int i = 0; i < 4; ++i) { const float d = v[i] - mu; s2 += d * d; }
    sred[tid] = s2;
    __syncthreads();
    for (int o = 128; o > 0; o >>= 1) {
        if (tid < o) sred[tid] += sred[tid + o];
        __syncthreads();
    }
    const float rstd = rsqrtf(sred[0] * (1.0f / DD) + 1e-5f);

#pragma unroll
    for (int i = 0; i < 4; ++i) {
        const int c = tid + 256 * i;
        xn[(size_t)t * DD + c] = (_Float16)((v[i] - mu) * rstd * gamma[c] + beta[c]);
    }
}

// ---------------------------------------------------------------------------
// Kernel 2: f32 -> f16 weight conversion (grid-stride)
// ---------------------------------------------------------------------------
__global__ __launch_bounds__(256) void wconv_kernel(const float* __restrict__ src,
                                                    _Float16* __restrict__ dst, int n) {
    for (int i = blockIdx.x * blockDim.x + threadIdx.x; i < n; i += gridDim.x * blockDim.x)
        dst[i] = (_Float16)src[i];
}

// ---------------------------------------------------------------------------
// Kernel 3: fused QKV GEMM.
// Wave = 16(token) x 64(feature == one head). Q,K -> [B,H,S,DH]; V -> [B,H,DH,S].
// ---------------------------------------------------------------------------
__global__ __launch_bounds__(256) void qkv_gemm_kernel(const _Float16* __restrict__ xn,
                                                       const _Float16* __restrict__ wall,
                                                       const float* __restrict__ bq,
                                                       const float* __restrict__ bk,
                                                       const float* __restrict__ bv,
                                                       _Float16* __restrict__ qo,
                                                       _Float16* __restrict__ ko,
                                                       _Float16* __restrict__ vT) {
    const int wid  = (blockIdx.x * blockDim.x + threadIdx.x) >> 5;
    const int lane = threadIdx.x & 31;
    const int lo16 = lane & 15;
    const int half = lane >> 4;

    const int mat = wid / (512 * HH);
    const int rem = wid % (512 * HH);
    const int ht  = rem / 512;
    const int tt  = rem % 512;

    const _Float16* W    = wall + (size_t)mat * DD * DD;
    const float*    bias = (mat == 0) ? bq : (mat == 1) ? bk : bv;

    const _Float16* arow = xn + (size_t)(tt * 16 + lo16) * DD;
    const _Float16* wrow[4];
#pragma unroll
    for (int g = 0; g < 4; ++g)
        wrow[g] = W + (size_t)(ht * DH + g * 16 + lo16) * DD;

    v8f acc[4] = {v8f{}, v8f{}, v8f{}, v8f{}};

    for (int kk = 0; kk < DD / 32; ++kk) {
        const int d0 = kk * 32;
        __builtin_prefetch(arow + d0 + 256, 0, 1);   // global_prefetch_b8
        const v8h alo = *(const v8h*)(arow + d0 + 8 * half);
        const v8h ahi = *(const v8h*)(arow + d0 + 8 * half + 16);
        const v16h a  = cat8(alo, ahi);
#pragma unroll
        for (int g = 0; g < 4; ++g) {
            const v16h bmat = cat8(*(const v8h*)(wrow[g] + d0 + 8 * half), *(const v8h*)(wrow[g] + d0 + 8 * half + 16));
            acc[g] = __builtin_amdgcn_wmma_f32_16x16x32_f16(
                false, a, false, bmat, (short)0, acc[g], false, false);
        }
    }

    const int b     = (tt * 16) >> 11;
    const int sbase = (tt * 16) & (SS - 1);
#pragma unroll
    for (int g = 0; g < 4; ++g) {
        const int   n  = ht * DH + g * 16 + lo16;
        const float bn = bias[n];
        const int   dh = g * 16 + lo16;
        if (mat == 2) {
            v8h pack;
#pragma unroll
            for (int r = 0; r < 8; ++r)
                pack[r] = (_Float16)(acc[g][r] + bn);
            const size_t idx = ((size_t)(b * HH + ht) * DH + dh) * SS + sbase + 8 * half;
            *(v8h*)(vT + idx) = pack;
        } else {
            _Float16* dst = (mat == 0) ? qo : ko;
#pragma unroll
            for (int r = 0; r < 8; ++r) {
                const int s = sbase + 8 * half + r;
                dst[((size_t)(b * HH + ht) * SS + s) * DH + dh] = (_Float16)(acc[g][r] + bn);
            }
        }
    }
}

// ---------------------------------------------------------------------------
// Kernel 4: cooperative flash attention.
// Block = 4 waves, one (b,h), 4 consecutive 16-query tiles (64 queries).
// Each 32-key chunk: K (32x64, 4KB contiguous) and V^T (64x32, 4KB) are
// async-copied into double-buffered LDS shared by all 4 waves
// (global_load_async_to_lds_b128 + s_wait_asynccnt), then each wave runs
// 4 WMMA (scores) + 4 WMMA (P@V) with online softmax.
// ---------------------------------------------------------------------------
__global__ __launch_bounds__(128) void attn_kernel(const _Float16* __restrict__ q,
                                                   const _Float16* __restrict__ k,
                                                   const _Float16* __restrict__ vT,
                                                   float* __restrict__ out) {
    __shared__ __align__(32) _Float16 Kc[2][32][DH];   // 8 KB: keys t-local x d
    __shared__ __align__(32) _Float16 Vc[2][DH][32];   // 8 KB: d x t-local (V^T)
    __shared__ __align__(32) _Float16 Pl[4][16][32];   // 4 KB: per-wave P transpose

    const int tid  = threadIdx.x;
    const int widx = tid >> 5;
    const int lane = tid & 31;
    const int lo16 = lane & 15;
    const int half = lane >> 4;

    const int bh = blockIdx.x >> 5;          // (b*H + h), 64 values
    const int qg = blockIdx.x & 31;          // 64-query group
    const int s0 = qg * 64 + widx * 16;      // this wave's query tile
    const int b  = bh >> 4;
    const int h  = bh & 15;

    const _Float16* Qb = q  + (size_t)bh * SS * DH;
    const _Float16* Kb = k  + (size_t)bh * SS * DH;
    const _Float16* Vb = vT + (size_t)bh * DH * SS;

    // cooperative chunk copy: 128 threads x 32B each for K and for V^T
    const int vrow = tid >> 1, vhh = tid & 1;
    auto issue_chunk = [&](int t0, int buf) {
        const _Float16* gk = Kb + (size_t)t0 * DH + tid * 16;   // contiguous 4KB slab
        _Float16*       lk = &Kc[buf][0][0] + tid * 16;
        cp_b128_to_lds(gk, lk);
        cp_b128_to_lds(gk + 8, lk + 8);
        const _Float16* gv = Vb + (size_t)vrow * SS + t0 + vhh * 16;
        _Float16*       lv = &Vc[buf][vrow][vhh * 16];
        cp_b128_to_lds(gv, lv);
        cp_b128_to_lds(gv + 8, lv + 8);
    };

    // Q tiles in A layout, resident in registers
    const _Float16* qrow = Qb + (size_t)(s0 + lo16) * DH;
    v16h qa[2];
#pragma unroll
    for (int t = 0; t < 2; ++t) {
        const int d = t * 32 + 8 * half;
        qa[t] = cat8(*(const v8h*)(qrow + d), *(const v8h*)(qrow + d + 16));
    }

    float mrun[8], lrun[8];
#pragma unroll
    for (int r = 0; r < 8; ++r) { mrun[r] = -INFINITY; lrun[r] = 0.f; }
    v8f acc[4] = {v8f{}, v8f{}, v8f{}, v8f{}};

    issue_chunk(0, 0);

    for (int t0 = 0; t0 < SS; t0 += 32) {
        const int buf  = (t0 >> 5) & 1;
        const int more = (t0 + 32 < SS);
        if (more) issue_chunk(t0 + 32, buf ^ 1);
        wait_async_le(more);     // chunk `buf` has landed (in-order completion)
        __syncthreads();

        // ---- scores: two 16x16 tiles over two K-steps, B operands from LDS
        v8f sc[2] = {v8f{}, v8f{}};
#pragma unroll
        for (int ng = 0; ng < 2; ++ng) {
            const _Float16* krow = &Kc[buf][ng * 16 + lo16][0];
            const v16h kb0 = cat8(*(const v8h*)(krow + 8 * half), *(const v8h*)(krow + 8 * half + 16));
            const v16h kb1 = cat8(*(const v8h*)(krow + 32 + 8 * half), *(const v8h*)(krow + 32 + 8 * half + 16));
            sc[ng] = __builtin_amdgcn_wmma_f32_16x16x32_f16(
                false, qa[0], false, kb0, (short)0, sc[ng], false, false);
            sc[ng] = __builtin_amdgcn_wmma_f32_16x16x32_f16(
                false, qa[1], false, kb1, (short)0, sc[ng], false, false);
        }

        // ---- online softmax (row m = r + 8*half, cols across 16 lanes)
        const float sscale = 0.125f; // 1/sqrt(64)
#pragma unroll
        for (int r = 0; r < 8; ++r) {
            const float x0 = sc[0][r] * sscale;
            const float x1 = sc[1][r] * sscale;
            float mx = fmaxf(x0, x1);
#pragma unroll
            for (int off = 1; off < 16; off <<= 1)
                mx = fmaxf(mx, __shfl_xor(mx, off, 32));
            const float mnew  = fmaxf(mrun[r], mx);
            const float alpha = __expf(mrun[r] - mnew);
            const float p0 = __expf(x0 - mnew);
            const float p1 = __expf(x1 - mnew);
            float rs = p0 + p1;
#pragma unroll
            for (int off = 1; off < 16; off <<= 1)
                rs += __shfl_xor(rs, off, 32);
            lrun[r] = lrun[r] * alpha + rs;
            mrun[r] = mnew;
#pragma unroll
            for (int g = 0; g < 4; ++g) acc[g][r] *= alpha;
            const int m = r + 8 * half;
            Pl[widx][m][lo16]      = (_Float16)p0;
            Pl[widx][m][16 + lo16] = (_Float16)p1;
        }

        asm volatile("s_wait_dscnt 0" ::: "memory");   // wave-private P slice

        // ---- P in A layout
        const v8h plo = *(const v8h*)(&Pl[widx][lo16][8 * half]);
        const v8h phi = *(const v8h*)(&Pl[widx][lo16][16 + 8 * half]);
        const v16h pa = cat8(plo, phi);

        // ---- ctx += P @ V, B operands from LDS (V^T rows contiguous in t)
#pragma unroll
        for (int g = 0; g < 4; ++g) {
            const v16h vb = cat8(*(const v8h*)(&Vc[buf][g * 16 + lo16][8 * half]), *(const v8h*)(&Vc[buf][g * 16 + lo16][16 + 8 * half]));
            acc[g] = __builtin_amdgcn_wmma_f32_16x16x32_f16(
                false, pa, false, vb, (short)0, acc[g], false, false);
        }
        __syncthreads();   // protect buffer `buf` before it is refilled
    }

    // ---- normalize + store [B,S,D] fp32
#pragma unroll
    for (int r = 0; r < 8; ++r) {
        const float inv = 1.0f / lrun[r];
        const int   s   = s0 + r + 8 * half;
#pragma unroll
        for (int g = 0; g < 4; ++g)
            out[((size_t)b * SS + s) * DD + h * DH + g * 16 + lo16] = acc[g][r] * inv;
    }
}

// ---------------------------------------------------------------------------
extern "C" void kernel_launch(void* const* d_in, const int* in_sizes, int n_in,
                              void* d_out, int out_size, void* d_ws, size_t ws_size,
                              hipStream_t stream) {
    (void)in_sizes; (void)n_in; (void)out_size; (void)ws_size;

    const float* hs    = (const float*)d_in[0];
    const float* gamma = (const float*)d_in[1];
    const float* beta  = (const float*)d_in[2];
    const float* Wq    = (const float*)d_in[3];
    const float* bq    = (const float*)d_in[4];
    const float* Wk    = (const float*)d_in[5];
    const float* bk    = (const float*)d_in[6];
    const float* Wv    = (const float*)d_in[7];
    const float* bv    = (const float*)d_in[8];

    char* ws = (char*)d_ws;
    _Float16* xn   = (_Float16*)(ws);                                   // 16 MB
    _Float16* wall = (_Float16*)(ws + (size_t)16 * 1024 * 1024);        //  6 MB
    _Float16* qbuf = (_Float16*)(ws + (size_t)22 * 1024 * 1024);        // 16 MB
    _Float16* kbuf = (_Float16*)(ws + (size_t)38 * 1024 * 1024);        // 16 MB
    _Float16* vTbf = (_Float16*)(ws + (size_t)54 * 1024 * 1024);        // 16 MB

    ln_kernel<<<NTOK, 256, 0, stream>>>(hs, gamma, beta, xn);

    const int WN = DD * DD;
    wconv_kernel<<<1024, 256, 0, stream>>>(Wq, wall, WN);
    wconv_kernel<<<1024, 256, 0, stream>>>(Wk, wall + (size_t)WN, WN);
    wconv_kernel<<<1024, 256, 0, stream>>>(Wv, wall + (size_t)2 * WN, WN);

    // 3 mats * 16 head-tiles * 512 token-tiles = 24576 waves, 8 waves/block
    qkv_gemm_kernel<<<3072, 256, 0, stream>>>(xn, wall, bq, bk, bv, qbuf, kbuf, vTbf);

    // 64 (b,h) * 32 query-groups = 2048 blocks of 4 waves
    attn_kernel<<<2048, 128, 0, stream>>>(qbuf, kbuf, vTbf, (float*)d_out);
}
